// Attention_22892175687737
// MI455X (gfx1250) — compile-verified, matrix operand order checked by reading, not run on hardware
//
#include <hip/hip_runtime.h>

// ---------------------------------------------------------------------------
// Types
// ---------------------------------------------------------------------------
typedef __attribute__((ext_vector_type(16))) __bf16    v16bf;
typedef __attribute__((ext_vector_type(8)))  __bf16    v8bf;
typedef __attribute__((ext_vector_type(8)))  float     v8f;
typedef __attribute__((ext_vector_type(4)))  float     v4f;
typedef __attribute__((ext_vector_type(4)))  unsigned  u32x4;
typedef __attribute__((ext_vector_type(8)))  unsigned  u32x8;

// Problem constants (from reference)
constexpr int BATCH    = 2;
constexpr int SEQ      = 2048;
constexpr int DIM      = 2048;
constexpr int NHEADS   = 16;
constexpr int HDIM     = 128;
constexpr int QKV_H    = NHEADS * 3;          // 48 head-slots per token
constexpr int QKV_OUT  = QKV_H * HDIM;        // 6144
constexpr int MROWS    = BATCH * SEQ;         // 4096

// ---------------------------------------------------------------------------
// Helpers
// ---------------------------------------------------------------------------
__device__ __forceinline__ __bf16 bf16_from_f32(float x) {
    unsigned u = __builtin_bit_cast(unsigned, x);
    unsigned r = u + 0x7FFFu + ((u >> 16) & 1u);       // round-to-nearest-even
    unsigned short h = (unsigned short)(r >> 16);
    return __builtin_bit_cast(__bf16, h);
}

__device__ __forceinline__ v8f wmma_bf16(v16bf a, v16bf b, v8f c) {
    // D = A(16x32) * B(32x16) + C, fp32 accumulate -> v_wmma_f32_16x16x32_bf16
    return __builtin_amdgcn_wmma_f32_16x16x32_bf16(
        false, a, false, b, (short)0, c, false, false);
}

// Low 32 bits of a generic pointer into LDS == LDS byte offset (aperture map)
__device__ __forceinline__ unsigned lds_bytes(const void* p) {
    return (unsigned)(unsigned long long)(uintptr_t)p;
}

// Load one 16x32 bf16 A/B fragment from LDS laid out row-major with `stride`
// (in bf16 elements). Per ISA 16-bit A-matrix layout (wave32):
//   lanes 0-15 : row M=lane,   elems 0..7 = K0..7,   8..15 = K16..23
//   lanes 16-31: row M=lane-16,elems 0..7 = K8..15,  8..15 = K24..31
__device__ __forceinline__ v16bf load_frag(const __bf16* base, int stride, int lane) {
    int r   = lane & 15;
    int grp = lane >> 4;
    const __bf16* p = base + r * stride + grp * 8;
    v8bf lo = *(const v8bf*)p;
    v8bf hi = *(const v8bf*)(p + 16);
    v16bf out;
#pragma unroll
    for (int i = 0; i < 8; ++i) { out[i] = lo[i]; out[i + 8] = hi[i]; }
    return out;
}

// Load 16 contiguous elements (fp32 or bf16) -> 16 bf16
template <typename T>
__device__ __forceinline__ void load16_bf(const T* src, __bf16* out) {
    if constexpr (sizeof(T) == 4) {
        const v4f* s4 = (const v4f*)src;
#pragma unroll
        for (int q = 0; q < 4; ++q) {
            v4f v = s4[q];
            out[q * 4 + 0] = bf16_from_f32(v.x);
            out[q * 4 + 1] = bf16_from_f32(v.y);
            out[q * 4 + 2] = bf16_from_f32(v.z);
            out[q * 4 + 3] = bf16_from_f32(v.w);
        }
    } else {
        v8bf a = ((const v8bf*)src)[0];
        v8bf b = ((const v8bf*)src)[1];
#pragma unroll
        for (int i = 0; i < 8; ++i) { out[i] = a[i]; out[8 + i] = b[i]; }
    }
}

template <typename T>
__device__ __forceinline__ void stage_row16(const T* src, __bf16* dst) {
    __bf16 t[16];
    load16_bf(src, t);
    v8bf lo, hi;
#pragma unroll
    for (int i = 0; i < 8; ++i) { lo[i] = t[i]; hi[i] = t[i + 8]; }
    ((v8bf*)dst)[0] = lo;
    ((v8bf*)dst)[1] = hi;
}

// ---------------------------------------------------------------------------
// TDM: DMA one 128x128 bf16 tile (global row stride = QKV_H*HDIM elements)
// into LDS with a padded row stride of LSTR = 136 bf16 (pad 4 DWORDs every
// 64 DWORDs).  D# group0/group1 per CDNA5 ISA.
// Issued by one wave; tracked with TENSORcnt.
// ---------------------------------------------------------------------------
__device__ __forceinline__ void tdm_load_tile_128x128(const __bf16* gsrc,
                                                      unsigned lds_byte) {
    unsigned long long ga = (unsigned long long)(uintptr_t)gsrc;
    u32x4 g0;
    g0[0] = 1u;                                         // count=1, user desc
    g0[1] = lds_byte;                                   // lds_addr
    g0[2] = (unsigned)ga;                               // global_addr[31:0]
    g0[3] = (unsigned)((ga >> 32) & 0x01FFFFFFull)      // global_addr[56:32]
          | (2u << 30);                                 // type=2 ("image")
    u32x8 g1;
    g1[0] = (1u << 16)      // data_size = 1 -> 2 bytes
          | (1u << 20)      // pad_enable
          | (5u << 22)      // pad_interval: 64 DWORDs (256B) before padding
          | (3u << 25);     // pad_amount:  4 DWORDs (16B)
    g1[1] = (128u << 16);   // [47:32]=atomic addr 0; [63:48]=tensor_dim0 lo = 128
    g1[2] = (128u << 16);   // [79:64]=tensor_dim0 hi 0; [95:80]... tensor_dim1 lo = 128
    g1[3] = (128u << 16);   // [111:96]=tensor_dim1 hi 0; [127:112]=tile_dim0 = 128
    g1[4] = 128u;           // [143:128]=tile_dim1 = 128; tile_dim2 = 0
    g1[5] = (unsigned)(QKV_H * HDIM);  // tensor_dim0_stride = 6144 (lo 32)
    g1[6] = 0u;             // stride hi, tensor_dim1_stride lo
    g1[7] = 0u;
    asm volatile("tensor_load_to_lds %0, %1" :: "s"(g0), "s"(g1) : "memory");
}

// ---------------------------------------------------------------------------
// Generic WMMA GEMM: C(MxN) = A(MxK) * B(KxN), row-major, sizes divide tiles.
// Block tile 128x128x32, 256 threads = 8 waves (2x4), wave tile 64x32.
// A and B converted to bf16 in LDS; both read back as identical 16x32 frags
// (B staged transposed so its per-lane fragment == a contiguous column run).
// bf16 A tiles are staged with async global->LDS copies (ASYNCcnt).
// ---------------------------------------------------------------------------
constexpr int GBM = 128, GBN = 128, GBK = 32, GPAD = 8, GSTR = GBK + GPAD;

template <typename TA, typename TB, typename TC>
__global__ __launch_bounds__(256) void gemm_wmma(const TA* __restrict__ A,
                                                 const TB* __restrict__ Bm,
                                                 TC* __restrict__ C,
                                                 int M, int N, int K) {
    __shared__ __bf16 As[GBM][GSTR];
    __shared__ __bf16 Bs[GBN][GSTR];   // stored transposed: Bs[n][k]

    const int tid   = threadIdx.x;
    const int lane  = tid & 31;
    const int wid   = tid >> 5;
    const int waveM = wid >> 2;        // 0..1
    const int waveN = wid & 3;         // 0..3
    const int mbase = blockIdx.y * GBM;
    const int nbase = blockIdx.x * GBN;

    v8f acc[4][2] = {};

    for (int k0 = 0; k0 < K; k0 += GBK) {
        // ---- stage A tile (128x32) ----
        {
            int r = tid >> 1, c = (tid & 1) * 16;
            const TA* srcA = &A[(size_t)(mbase + r) * K + k0 + c];
            if constexpr (sizeof(TA) == 2) {
                // bf16 A: pure copy -> async global->LDS (32B per thread),
                // IOFFSET applies to both global and LDS addresses.
                unsigned dstA = lds_bytes(&As[r][c]);
                unsigned long long ga = (unsigned long long)(uintptr_t)srcA;
                asm volatile("global_load_async_to_lds_b128 %0, %1, off"
                             :: "v"(dstA), "v"(ga) : "memory");
                asm volatile("global_load_async_to_lds_b128 %0, %1, off offset:16"
                             :: "v"(dstA), "v"(ga) : "memory");
            } else {
                stage_row16(srcA, &As[r][c]);
            }
        }
        // ---- stage B tile (32x128), transposed into Bs[n][k] ----
        {
            int kk = tid >> 3, nb = (tid & 7) * 16;
            __bf16 t[16];
            load16_bf(&Bm[(size_t)(k0 + kk) * N + nbase + nb], t);
#pragma unroll
            for (int i = 0; i < 16; ++i) Bs[nb + i][kk] = t[i];
        }
        // prefetch next K tile (global_prefetch_b8)
        if (k0 + GBK < K) {
            int r = tid >> 1;
            __builtin_prefetch(&A[(size_t)(mbase + r) * K + k0 + GBK], 0, 0);
            __builtin_prefetch(&Bm[(size_t)(k0 + GBK + (tid >> 3)) * N + nbase], 0, 0);
        }
        if constexpr (sizeof(TA) == 2) {
            asm volatile("s_wait_asynccnt 0x0" ::: "memory");
        }
        __syncthreads();

        v16bf af[4], bfr[2];
#pragma unroll
        for (int i = 0; i < 4; ++i)
            af[i] = load_frag(&As[waveM * 64 + i * 16][0], GSTR, lane);
#pragma unroll
        for (int j = 0; j < 2; ++j)
            bfr[j] = load_frag(&Bs[waveN * 32 + j * 16][0], GSTR, lane);
#pragma unroll
        for (int i = 0; i < 4; ++i)
#pragma unroll
            for (int j = 0; j < 2; ++j)
                acc[i][j] = wmma_bf16(af[i], bfr[j], acc[i][j]);
        __syncthreads();
    }

    // ---- epilogue: C-fragment layout M = v + 8*grp, N = lane&15 ----
    const int grp = lane >> 4, lcol = lane & 15;
#pragma unroll
    for (int i = 0; i < 4; ++i)
#pragma unroll
        for (int j = 0; j < 2; ++j) {
            size_t row0 = (size_t)mbase + waveM * 64 + i * 16 + grp * 8;
            size_t col  = (size_t)nbase + waveN * 32 + j * 16 + lcol;
#pragma unroll
            for (int v = 0; v < 8; ++v) {
                float val = acc[i][j][v];
                if constexpr (sizeof(TC) == 4) C[(row0 + v) * N + col] = val;
                else                           C[(row0 + v) * N + col] = bf16_from_f32(val);
            }
        }
}

// ---------------------------------------------------------------------------
// Flash attention: one block per (b, head, 128-row Q tile). 8 waves; each wave
// owns 16 Q rows. Q/K tiles staged by the Tensor Data Mover (wave 0 issues the
// descriptor, EXEC is ignored by TDM); V staged transposed by all threads.
// qkv layout: [b][s][48 heads][128]; heads 0..15=Q, 16..31=K, 32..47=V.
// Output: combined bf16 [b][s][16*128].
// ---------------------------------------------------------------------------
constexpr int LSTR = HDIM + 8;   // padded LDS row stride (bf16 elems) == TDM pad

__global__ __launch_bounds__(256) void attn_kernel(const __bf16* __restrict__ qkv,
                                                   __bf16* __restrict__ outc) {
    extern __shared__ __bf16 smem[];
    __bf16* KT = smem;                 // K tile [128][LSTR]; later aliased by P
    __bf16* VT = smem + 128 * LSTR;    // V^T tile [d=128][LSTR] (n along rows)

    const int tid  = threadIdx.x;
    const int lane = tid & 31;
    const int wid  = tid >> 5;
    const int grp  = lane >> 4;
    const int lcol = lane & 15;
    const int wrow = wid * 16;

    const int bh = blockIdx.y;           // 0..31
    const int b  = bh >> 4, h = bh & 15;
    const int qbase = blockIdx.x * 128;

    const size_t rstride = (size_t)QKV_H * HDIM;  // 6144
    const __bf16* Qg = qkv + (size_t)b * SEQ * rstride + (size_t)h * HDIM;
    const __bf16* Kg = Qg + (size_t)NHEADS * HDIM;
    const __bf16* Vg = Qg + (size_t)2 * NHEADS * HDIM;

    // ---- phase 0: TDM-stage Q tile, keep as 4 A-fragments ----
    if (wid == 0) {
        tdm_load_tile_128x128(Qg + (size_t)qbase * rstride, lds_bytes(KT));
        __builtin_amdgcn_s_wait_tensorcnt(0);
    }
    __syncthreads();
    v16bf qf[4];
#pragma unroll
    for (int ks = 0; ks < 4; ++ks)
        qf[ks] = load_frag(&KT[wrow * LSTR + ks * 32], LSTR, lane);
    __syncthreads();

    v8f acc[8] = {};
    float mrow[8], srow[8];
#pragma unroll
    for (int j = 0; j < 8; ++j) { mrow[j] = -1e30f; srow[j] = 0.f; }
    const float qscale = 0.08838834764831845f;   // 1/sqrt(128)

    for (int t = 0; t < SEQ / 128; ++t) {
        const int kb = t * 128;
        // ---- TDM-stage K tile [s][d] (wave 0) ----
        if (wid == 0)
            tdm_load_tile_128x128(Kg + (size_t)kb * rstride, lds_bytes(KT));
        // ---- stage V transposed: VT[d][n] (all threads) ----
        {
            int n = tid >> 1, d0 = (tid & 1) * 64;
            const __bf16* src = Vg + (size_t)(kb + n) * rstride + d0;
#pragma unroll
            for (int i = 0; i < 8; ++i) {
                v8bf v = *(const v8bf*)(src + i * 8);
#pragma unroll
                for (int u = 0; u < 8; ++u)
                    VT[(d0 + i * 8 + u) * LSTR + n] = v[u];
            }
        }
        if (wid == 0) __builtin_amdgcn_s_wait_tensorcnt(0);
        __syncthreads();

        // ---- logits: S = Q * K^T  (B-frag of K^T == contiguous K row) ----
        v8f lg[8];
#pragma unroll
        for (int nf = 0; nf < 8; ++nf) {
            v8f c = {};
#pragma unroll
            for (int ks = 0; ks < 4; ++ks) {
                v16bf kf = load_frag(&KT[nf * 16 * LSTR + ks * 32], LSTR, lane);
                c = wmma_bf16(qf[ks], kf, c);
            }
            lg[nf] = c;
        }
#pragma unroll
        for (int nf = 0; nf < 8; ++nf)
#pragma unroll
            for (int j = 0; j < 8; ++j) lg[nf][j] *= qscale;

        // ---- online softmax: per-row (j,grp) stats across 16-lane half ----
        float scl[8];
#pragma unroll
        for (int j = 0; j < 8; ++j) {
            float pm = -1e30f;
#pragma unroll
            for (int nf = 0; nf < 8; ++nf) pm = fmaxf(pm, lg[nf][j]);
#pragma unroll
            for (int o = 1; o < 16; o <<= 1) pm = fmaxf(pm, __shfl_xor(pm, o, 32));
            float mnew = fmaxf(mrow[j], pm);
            scl[j] = __expf(mrow[j] - mnew);
            mrow[j] = mnew;
        }
#pragma unroll
        for (int j = 0; j < 8; ++j) {
            float ps = 0.f;
#pragma unroll
            for (int nf = 0; nf < 8; ++nf) {
                float p = __expf(lg[nf][j] - mrow[j]);
                lg[nf][j] = p;
                ps += p;
            }
#pragma unroll
            for (int o = 1; o < 16; o <<= 1) ps += __shfl_xor(ps, o, 32);
            srow[j] = srow[j] * scl[j] + ps;
#pragma unroll
            for (int df = 0; df < 8; ++df) acc[df][j] *= scl[j];
        }

        // ---- P (C-layout f32) -> LDS bf16 (aliasing K region) -> A-frags ----
        __syncthreads();   // everyone done reading KT
        __bf16* P = KT + wrow * LSTR;     // this wave's 16 rows
#pragma unroll
        for (int nf = 0; nf < 8; ++nf)
#pragma unroll
            for (int j = 0; j < 8; ++j)
                P[(j + grp * 8) * LSTR + nf * 16 + lcol] = bf16_from_f32(lg[nf][j]);

        v16bf pf[4];
#pragma unroll
        for (int ks = 0; ks < 4; ++ks) pf[ks] = load_frag(P + ks * 32, LSTR, lane);

        // ---- O += P * V  (B-frag of V == contiguous V^T row) ----
#pragma unroll
        for (int df = 0; df < 8; ++df)
#pragma unroll
            for (int ks = 0; ks < 4; ++ks) {
                v16bf vf = load_frag(&VT[df * 16 * LSTR + ks * 32], LSTR, lane);
                acc[df] = wmma_bf16(pf[ks], vf, acc[df]);
            }
        __syncthreads();   // before restaging K/V next iteration
    }

    // ---- epilogue: normalize, store combined[b][s][h*128 + d] ----
    __bf16* dst = outc + ((size_t)b * SEQ + qbase + wrow) * (NHEADS * HDIM)
                       + (size_t)h * HDIM;
#pragma unroll
    for (int df = 0; df < 8; ++df)
#pragma unroll
        for (int j = 0; j < 8; ++j) {
            float v = acc[df][j] / srow[j];
            dst[(size_t)(j + grp * 8) * (NHEADS * HDIM) + df * 16 + lcol] =
                bf16_from_f32(v);
        }
}

// ---------------------------------------------------------------------------
// Host launcher
// ---------------------------------------------------------------------------
extern "C" void kernel_launch(void* const* d_in, const int* in_sizes, int n_in,
                              void* d_out, int out_size, void* d_ws, size_t ws_size,
                              hipStream_t stream) {
    (void)in_sizes; (void)n_in; (void)out_size; (void)ws_size;

    const float* x     = (const float*)d_in[0];   // [2,2048,2048]
    const float* w_qkv = (const float*)d_in[1];   // [2048,6144]
    const float* w_out = (const float*)d_in[2];   // [2048,2048]
    float*       out   = (float*)d_out;           // [2,2048,2048]

    // Workspace: bf16 qkv (50.3 MB) + bf16 combined (16.8 MB)
    __bf16* qkv  = (__bf16*)d_ws;
    __bf16* comb = (__bf16*)((char*)d_ws +
                             (size_t)MROWS * QKV_OUT * sizeof(__bf16));

    // 1) qkv = x @ w_qkv   (4096 x 6144 x 2048)
    gemm_wmma<float, float, __bf16>
        <<<dim3(QKV_OUT / GBN, MROWS / GBM), 256, 0, stream>>>(
            x, w_qkv, qkv, MROWS, QKV_OUT, DIM);

    // 2) flash attention per (b, head, q-tile)
    const int attn_smem = 2 * 128 * LSTR * (int)sizeof(__bf16);   // ~68 KB
    hipFuncSetAttribute((const void*)attn_kernel,
                        hipFuncAttributeMaxDynamicSharedMemorySize, attn_smem);
    attn_kernel<<<dim3(SEQ / 128, BATCH * NHEADS), 256, attn_smem, stream>>>(
        qkv, comb);

    // 3) out = combined @ w_out   (4096 x 2048 x 2048)
    gemm_wmma<__bf16, float, float>
        <<<dim3(DIM / GBN, MROWS / GBM), 256, 0, stream>>>(
            comb, w_out, out, MROWS, DIM, DIM);
}
